// MultiHeadAttention_4887672783317
// MI455X (gfx1250) — matrix operand order repaired and checked by reading, not compiled, not run on hardware
//
#include <hip/hip_runtime.h>

// MHA forward for B=4, S=2048, D=512, H=8, DK=64 on gfx1250 (wave32, WMMA).
// d_out = [out (B*S*D fp32)] ++ [attn (B*H*S*S fp32)]
//
// GEMMs use v_wmma_f32_16x16x32_f16 with register blocking: each wave owns a
// 16(M) x 64(N) strip -> one A fragment is reused across 4 B fragments per
// 32-deep K step (4 independent accumulators, no WMMA->WMMA RAW hazards).

#define DFEAT 512
#define DK    64
#define NH    8
#define SEQ   2048

typedef __attribute__((ext_vector_type(16))) _Float16 v16h;
typedef __attribute__((ext_vector_type(8)))  _Float16 v8h;
typedef __attribute__((ext_vector_type(8)))  float    v8f;

// ---------------- WMMA fragment helpers (wave32, 16x16x32 f16) ----------------
// A (16x32 f16, MxK): lane l -> m=l&15, g=l>>4; halves j=0..7 -> K=8g+j,
// j=8..15 -> K=16+8g+(j-8). Two contiguous 16B chunks per lane.
__device__ __forceinline__ v16h load_a_frag_f16(const _Float16* A, int lda,
                                                int m0, int k0, int lane) {
  int m = lane & 15, g = lane >> 4;
  const _Float16* p = A + (size_t)(m0 + m) * lda + k0 + 8 * g;
  v8h lo = *(const v8h*)p;
  v8h hi = *(const v8h*)(p + 16);
  v16h f;
#pragma unroll
  for (int j = 0; j < 8; ++j) { f[j] = lo[j]; f[j + 8] = hi[j]; }
  return f;
}

// Same A fragment but source is fp32 (convert on the fly, fp32->f16 RNE).
__device__ __forceinline__ v16h load_a_frag_f32(const float* A, int lda,
                                                int m0, int k0, int lane) {
  int m = lane & 15, g = lane >> 4;
  const float* p = A + (size_t)(m0 + m) * lda + k0 + 8 * g;
  v8f lo = *(const v8f*)p;
  v8f hi = *(const v8f*)(p + 16);
  v16h f;
#pragma unroll
  for (int j = 0; j < 8; ++j) { f[j] = (_Float16)lo[j]; f[j + 8] = (_Float16)hi[j]; }
  return f;
}

// B (32x16 f16, KxN) where source Wt is "transposed" row-major [N, ldw] and
// B(k,n) = Wt[(n0+n)*ldw + k0 + k]. Lane l -> n=l&15, g=l>>4; halves j=0..7 ->
// K = 8g + j, j=8..15 -> K = 16+8g+(j-8)  =>  two 16B chunks per lane.
__device__ __forceinline__ v16h load_b_frag_T(const _Float16* Wt, int ldw,
                                              int n0, int k0, int lane) {
  int n = lane & 15, g = lane >> 4;
  const _Float16* p = Wt + (size_t)(n0 + n) * ldw + k0 + 8 * g;
  v8h lo = *(const v8h*)p;
  v8h hi = *(const v8h*)(p + 16);
  v16h f;
#pragma unroll
  for (int j = 0; j < 8; ++j) { f[j] = lo[j]; f[j + 8] = hi[j]; }
  return f;
}

__device__ __forceinline__ v8f wmma_f16(v16h a, v16h b, v8f c) {
  // (neg_a, A, neg_b, B, c_mod, C, reuse_a, reuse_b)
  return __builtin_amdgcn_wmma_f32_16x16x32_f16(false, a, false, b, (short)0, c,
                                                false, false);
}

// ---------------- elementwise fp32 -> f16 convert ----------------
__global__ void f32_to_f16_kernel(const float* __restrict__ x,
                                  _Float16* __restrict__ y, int n) {
  int i = blockIdx.x * blockDim.x + threadIdx.x;
  if (i < n) y[i] = (_Float16)x[i];
}

// ---------------- QKV projection: Y = X @ W^T + b ----------------
// X fp32 [BS, 512] (b-major rows), Wh f16 [512, 512] (out,in), bias fp32[512].
// Each wave: 16 x 64 strip.  mode 0: out f16 [B,H,S,DK]; mode 1 (for V):
// out f16 [B,H,DK,S] (transposed for contiguous-K B fragments later).
__global__ void proj_kernel(const float* __restrict__ X,
                            const _Float16* __restrict__ Wh,
                            const float* __restrict__ bias,
                            _Float16* __restrict__ out, int mode) {
  int wave = threadIdx.x >> 5, lane = threadIdx.x & 31;
  int m0 = blockIdx.x * 16;
  int n0 = (blockIdx.y * 4 + wave) * 64;       // 64-wide strip
  int n = lane & 15, g = lane >> 4;
  v8f acc[4];
#pragma unroll
  for (int t = 0; t < 4; ++t) {
    float bval = bias[n0 + 16 * t + n];
    acc[t] = (v8f){bval, bval, bval, bval, bval, bval, bval, bval};
  }
  for (int k0 = 0; k0 < DFEAT; k0 += 32) {
    v16h a = load_a_frag_f32(X, DFEAT, m0, k0, lane);
#pragma unroll
    for (int t = 0; t < 4; ++t) {
      v16h b = load_b_frag_T(Wh, DFEAT, n0 + 16 * t, k0, lane);
      acc[t] = wmma_f16(a, b, acc[t]);
    }
  }
#pragma unroll
  for (int t = 0; t < 4; ++t) {
    int ng = n0 + 16 * t + n;
    int h = ng >> 6;                 // head
    int d = ng & (DK - 1);           // dim within head
#pragma unroll
    for (int r = 0; r < 8; ++r) {    // D frag: M = r + 8g, N = lane&15
      int row = m0 + 8 * g + r;
      int bidx = row >> 11;          // / SEQ
      int s = row & (SEQ - 1);
      size_t idx;
      if (mode == 0)
        idx = ((size_t)(bidx * NH + h) * SEQ + s) * DK + d;
      else
        idx = ((size_t)(bidx * NH + h) * DK + d) * SEQ + s;
      out[idx] = (_Float16)acc[t][r];
    }
  }
}

// ---------------- scores = q @ k^T per (b,h), fp32 out ----------------
// Each wave: 16 x 64 strip of the SxS score matrix (A fragment reused 4x).
__global__ void scores_kernel(const _Float16* __restrict__ qh,
                              const _Float16* __restrict__ kh,
                              float* __restrict__ attn) {
  int bh = blockIdx.z;
  const _Float16* qb = qh + (size_t)bh * SEQ * DK;
  const _Float16* kb = kh + (size_t)bh * SEQ * DK;
  float* ab = attn + (size_t)bh * SEQ * SEQ;
  int wave = threadIdx.x >> 5, lane = threadIdx.x & 31;
  int m0 = blockIdx.x * 16;
  int n0 = (blockIdx.y * 4 + wave) * 64;
  v8f acc[4] = {};
#pragma unroll
  for (int k0 = 0; k0 < DK; k0 += 32) {
    v16h a = load_a_frag_f16(qb, DK, m0, k0, lane);
#pragma unroll
    for (int t = 0; t < 4; ++t) {
      v16h b = load_b_frag_T(kb, DK, n0 + 16 * t, k0, lane);  // B(k,n)=k[n,k]
      acc[t] = wmma_f16(a, b, acc[t]);
    }
  }
  int n = lane & 15, g = lane >> 4;
#pragma unroll
  for (int t = 0; t < 4; ++t)
#pragma unroll
    for (int r = 0; r < 8; ++r)
      ab[(size_t)(m0 + 8 * g + r) * SEQ + n0 + 16 * t + n] = acc[t][r];
}

// ---------------- row softmax over SEQ cols (one block per row) ----------------
__global__ void softmax_kernel(float* __restrict__ attn) {
  __shared__ float red[256];
  float* row = attn + (size_t)blockIdx.x * SEQ;
  int tid = threadIdx.x;
  float v[8];
  float m = -3.402823466e38f;
#pragma unroll
  for (int i = 0; i < 8; ++i) { v[i] = row[tid + 256 * i]; m = fmaxf(m, v[i]); }
  red[tid] = m; __syncthreads();
  for (int s = 128; s > 0; s >>= 1) {
    if (tid < s) red[tid] = fmaxf(red[tid], red[tid + s]);
    __syncthreads();
  }
  float mx = red[0]; __syncthreads();
  float sum = 0.f;
#pragma unroll
  for (int i = 0; i < 8; ++i) { v[i] = __expf(v[i] - mx); sum += v[i]; }
  red[tid] = sum; __syncthreads();
  for (int s = 128; s > 0; s >>= 1) {
    if (tid < s) red[tid] += red[tid + s];
    __syncthreads();
  }
  float inv = 1.f / red[0];
#pragma unroll
  for (int i = 0; i < 8; ++i) row[tid + 256 * i] = v[i] * inv;
}

// ---------------- context = attn @ v per (b,h) -> ctx f16 [B,S,H*DK] ----------
// Each wave covers the FULL DK=64 width -> each fp32 attn fragment is loaded
// exactly once.  Block's 4 waves take 4 consecutive M tiles.
__global__ void context_kernel(const float* __restrict__ attn,
                               const _Float16* __restrict__ vT,
                               _Float16* __restrict__ ctx) {
  int bh = blockIdx.z;
  int bidx = bh >> 3, h = bh & 7;
  const float* ab = attn + (size_t)bh * SEQ * SEQ;
  const _Float16* vb = vT + (size_t)bh * DK * SEQ;  // [DK, SEQ]
  int wave = threadIdx.x >> 5, lane = threadIdx.x & 31;
  int m0 = (blockIdx.x * 4 + wave) * 16;
  v8f acc[4] = {};
  for (int k0 = 0; k0 < SEQ; k0 += 32) {
    v16h a = load_a_frag_f32(ab, SEQ, m0, k0, lane); // attn fp32 -> f16
#pragma unroll
    for (int t = 0; t < 4; ++t) {
      v16h b = load_b_frag_T(vb, SEQ, 16 * t, k0, lane); // B(k,n)=vT[n,k]=v[k,n]
      acc[t] = wmma_f16(a, b, acc[t]);
    }
  }
  int n = lane & 15, g = lane >> 4;
#pragma unroll
  for (int t = 0; t < 4; ++t)
#pragma unroll
    for (int r = 0; r < 8; ++r) {
      int s = m0 + 8 * g + r;
      ctx[(size_t)(bidx * SEQ + s) * (NH * DK) + h * DK + 16 * t + n] =
          (_Float16)acc[t][r];
    }
}

// ---------------- out-proj: proj = ctx @ Wo^T + bo (fp32 out) ----------------
__global__ void outproj_kernel(const _Float16* __restrict__ ctx,
                               const _Float16* __restrict__ Wh,
                               const float* __restrict__ bias,
                               float* __restrict__ out) {
  int wave = threadIdx.x >> 5, lane = threadIdx.x & 31;
  int m0 = blockIdx.x * 16;
  int n0 = (blockIdx.y * 4 + wave) * 64;
  int n = lane & 15, g = lane >> 4;
  v8f acc[4];
#pragma unroll
  for (int t = 0; t < 4; ++t) {
    float bval = bias[n0 + 16 * t + n];
    acc[t] = (v8f){bval, bval, bval, bval, bval, bval, bval, bval};
  }
  for (int k0 = 0; k0 < DFEAT; k0 += 32) {
    v16h a = load_a_frag_f16(ctx, DFEAT, m0, k0, lane);
#pragma unroll
    for (int t = 0; t < 4; ++t) {
      v16h b = load_b_frag_T(Wh, DFEAT, n0 + 16 * t, k0, lane);
      acc[t] = wmma_f16(a, b, acc[t]);
    }
  }
#pragma unroll
  for (int t = 0; t < 4; ++t)
#pragma unroll
    for (int r = 0; r < 8; ++r)
      out[(size_t)(m0 + 8 * g + r) * DFEAT + n0 + 16 * t + n] = acc[t][r];
}

// ---------------- residual add + LayerNorm over D=512 ----------------
__global__ void ln_kernel(const float* __restrict__ proj,
                          const float* __restrict__ resid,
                          const float* __restrict__ gamma,
                          const float* __restrict__ beta,
                          float* __restrict__ out) {
  __shared__ float red[256];
  int row = blockIdx.x, tid = threadIdx.x;
  const float* p = proj + (size_t)row * DFEAT;
  const float* q = resid + (size_t)row * DFEAT;
  float x0 = p[tid] + q[tid];
  float x1 = p[tid + 256] + q[tid + 256];
  red[tid] = x0 + x1; __syncthreads();
  for (int s = 128; s > 0; s >>= 1) {
    if (tid < s) red[tid] += red[tid + s];
    __syncthreads();
  }
  float mu = red[0] * (1.0f / DFEAT); __syncthreads();
  float d0 = x0 - mu, d1 = x1 - mu;
  red[tid] = d0 * d0 + d1 * d1; __syncthreads();
  for (int s = 128; s > 0; s >>= 1) {
    if (tid < s) red[tid] += red[tid + s];
    __syncthreads();
  }
  float inv = rsqrtf(red[0] * (1.0f / DFEAT) + 1e-5f);
  out[(size_t)row * DFEAT + tid] = d0 * inv * gamma[tid] + beta[tid];
  out[(size_t)row * DFEAT + tid + 256] = d1 * inv * gamma[tid + 256] + beta[tid + 256];
}

extern "C" void kernel_launch(void* const* d_in, const int* in_sizes, int n_in,
                              void* d_out, int out_size, void* d_ws, size_t ws_size,
                              hipStream_t stream) {
  const float* Q  = (const float*)d_in[0];
  const float* K  = (const float*)d_in[1];
  const float* V  = (const float*)d_in[2];
  const float* Wq = (const float*)d_in[3];
  const float* bq = (const float*)d_in[4];
  const float* Wk = (const float*)d_in[5];
  const float* bk = (const float*)d_in[6];
  const float* Wv = (const float*)d_in[7];
  const float* bv = (const float*)d_in[8];
  const float* Wo = (const float*)d_in[9];
  const float* bo = (const float*)d_in[10];
  const float* gamma = (const float*)d_in[11];
  const float* beta  = (const float*)d_in[12];

  const int BS = in_sizes[0] / DFEAT;   // B * S = 8192
  const int B  = BS / SEQ;              // 4

  float* outp = (float*)d_out;                       // [BS, 512]
  float* attn = outp + (size_t)BS * DFEAT;           // [B, H, S, S]

  // workspace layout
  char* ws = (char*)d_ws;
  size_t o = 0;
  const size_t Wsz   = (size_t)DFEAT * DFEAT * sizeof(_Float16);   // 512 KB
  const size_t qkvsz = (size_t)BS * DFEAT * sizeof(_Float16);      // 8 MB
  _Float16* Wqh = (_Float16*)(ws + o); o += Wsz;
  _Float16* Wkh = (_Float16*)(ws + o); o += Wsz;
  _Float16* Wvh = (_Float16*)(ws + o); o += Wsz;
  _Float16* Woh = (_Float16*)(ws + o); o += Wsz;
  _Float16* qh  = (_Float16*)(ws + o); o += qkvsz;   // [B,H,S,DK]
  _Float16* kh  = (_Float16*)(ws + o); o += qkvsz;   // [B,H,S,DK]
  _Float16* vTh = (_Float16*)(ws + o); o += qkvsz;   // [B,H,DK,S]
  _Float16* ctx = (_Float16*)(ws + o); o += qkvsz;   // [B,S,H*DK]
  float*    prj = (float*)(ws + o);                  // [BS, 512] fp32

  // 1) weights -> f16
  const int wn = DFEAT * DFEAT;
  f32_to_f16_kernel<<<(wn + 255) / 256, 256, 0, stream>>>(Wq, Wqh, wn);
  f32_to_f16_kernel<<<(wn + 255) / 256, 256, 0, stream>>>(Wk, Wkh, wn);
  f32_to_f16_kernel<<<(wn + 255) / 256, 256, 0, stream>>>(Wv, Wvh, wn);
  f32_to_f16_kernel<<<(wn + 255) / 256, 256, 0, stream>>>(Wo, Woh, wn);

  // 2) Q/K/V projections: 4 waves/block, each wave a 16x64 strip
  dim3 pgrid(BS / 16, DFEAT / 256);                  // (512, 2)
  proj_kernel<<<pgrid, 128, 0, stream>>>(Q, Wqh, bq, qh, 0);
  proj_kernel<<<pgrid, 128, 0, stream>>>(K, Wkh, bk, kh, 0);
  proj_kernel<<<pgrid, 128, 0, stream>>>(V, Wvh, bv, vTh, 1);

  // 3) scores -> attn region of d_out
  dim3 sgrid(SEQ / 16, SEQ / 256, B * NH);           // (128, 8, 32)
  scores_kernel<<<sgrid, 128, 0, stream>>>(qh, kh, attn);

  // 4) softmax in place
  softmax_kernel<<<B * NH * SEQ, 256, 0, stream>>>(attn);

  // 5) context = attn @ v  (each wave: 16 rows x full DK=64)
  dim3 cgrid(SEQ / 64, 1, B * NH);                   // (32, 1, 32)
  context_kernel<<<cgrid, 128, 0, stream>>>(attn, vTh, ctx);

  // 6) output projection, then residual + layernorm
  dim3 ogrid(BS / 16, DFEAT / 256);                  // (512, 2)
  outproj_kernel<<<ogrid, 128, 0, stream>>>(ctx, Woh, bo, prj);
  ln_kernel<<<BS, 256, 0, stream>>>(prj, Q, gamma, beta, outp);
}
